// MHAttention_33990371180665
// MI455X (gfx1250) — compile-verified, matrix operand order checked by reading, not run on hardware
//
#include <hip/hip_runtime.h>
#include <cstdint>
#include <cstddef>

// ---------------- types & constants ----------------
typedef __bf16 bf16_t;
typedef __attribute__((ext_vector_type(16))) __bf16 v16bf;
typedef __attribute__((ext_vector_type(8)))  float  v8f;

#define B_DIM  16
#define S_DIM  2048
#define D_IN   512
#define H_DIM  128
#define D_OUT  512
#define N_PROJ 768   // 128 (Q) + 128 (K) + 512 (V) output columns
#define KCHUNK 128   // keys processed per attention iteration

union BF16Frag {
  v16bf v;
  uint4 q[2];
};

// A/B fragment for v_wmma_f32_16x16x32_bf16 when the lane's 32 K-elements are
// contiguous in memory. ISA layout (16-bit, 16x32): lanes 0-15 hold
// K = dk+{0..7, 16..23}, lanes 16-31 hold K = dk+{8..15, 24..31}.
__device__ __forceinline__ v16bf load_frag_contig(const bf16_t* rowptr, int dk, int lane_hi) {
  const int o0 = dk + (lane_hi ? 8 : 0);
  BF16Frag f;
  f.q[0] = *(const uint4*)(rowptr + o0);        // 8 bf16, 16B aligned
  f.q[1] = *(const uint4*)(rowptr + o0 + 16);   // 8 bf16
  return f.v;
}

// load 4 K-step fragments (K = 0..127 of a contiguous row) into distinct regs
__device__ __forceinline__ void load_frag4(v16bf o[4], const bf16_t* rowptr, int lane_hi) {
#pragma unroll
  for (int kk = 0; kk < 4; ++kk) o[kk] = load_frag_contig(rowptr, kk * 32, lane_hi);
}

__device__ __forceinline__ v8f wmma_bf16(v16bf a, v16bf b, v8f c) {
  return __builtin_amdgcn_wmma_f32_16x16x32_bf16(false, a, false, b, (short)0, c, false, false);
}

// ---------------- kernel 0: pack weights fp32 -> bf16 [768 x 512] ----------------
__global__ void mha_cvt_w(const float* __restrict__ Wq, const float* __restrict__ Wk,
                          const float* __restrict__ Wv, bf16_t* __restrict__ Wall) {
  int i = blockIdx.x * 256 + threadIdx.x;              // 768*512 = 393216 total
  if (i >= N_PROJ * D_IN) return;
  float v;
  if (i < H_DIM * D_IN)               v = Wq[i];
  else if (i < 2 * H_DIM * D_IN)      v = Wk[i - H_DIM * D_IN];
  else                                v = Wv[i - 2 * H_DIM * D_IN];
  Wall[i] = (bf16_t)v;
}

// ---------------- kernel 1: LayerNorm + QKV projection (bf16 WMMA) ----------------
// grid (S/16, B), block 256 (8 waves). Each block: 16 sequence rows.
// Q, K stored row-major [b][s][h]; V stored TRANSPOSED: Vt[b][col][s] so the
// attention PV B-fragments are contiguous along the key dimension.
__global__ void __launch_bounds__(256, 1)
mha_ln_proj(const float* __restrict__ x,
            const float* __restrict__ ln_w, const float* __restrict__ ln_b,
            const bf16_t* __restrict__ Wall,
            bf16_t* __restrict__ Q, bf16_t* __restrict__ K,
            bf16_t* __restrict__ Vt) {
  __shared__ __align__(16) bf16_t xn[16][D_IN];   // 16 KB normalized activations

  const int b  = blockIdx.y;
  const int m0 = blockIdx.x * 16;
  const int t  = threadIdx.x;

  // ---- fp32 LayerNorm: 16 threads per row, shuffle-reduce inside 16-lane group ----
  {
    const int row = t >> 4;          // 0..15
    const int i   = t & 15;
    const float* xr = x + ((size_t)(b * S_DIM + m0 + row)) * D_IN;
    float s = 0.f, s2 = 0.f;
#pragma unroll 8
    for (int c = i; c < D_IN; c += 16) { float v = xr[c]; s += v; s2 += v * v; }
#pragma unroll
    for (int off = 1; off < 16; off <<= 1) {
      s  += __shfl_xor(s,  off, 32);
      s2 += __shfl_xor(s2, off, 32);
    }
    const float mean = s * (1.0f / D_IN);
    const float var  = s2 * (1.0f / D_IN) - mean * mean;
    const float rstd = rsqrtf(var + 1e-5f);
#pragma unroll 8
    for (int c = i; c < D_IN; c += 16) {
      float v = (xr[c] - mean) * rstd * ln_w[c] + ln_b[c];
      xn[row][c] = (bf16_t)v;
    }
  }
  __syncthreads();

  // ---- projection: y = xn @ Wall^T. Each wave owns 6 column tiles, processed
  // in TWO passes of 3 tiles to keep VGPR pressure low (no spills) while still
  // double-buffering the B fragments (b0/b1) against the 3-WMMA chains.
  const int wave    = t >> 5;
  const int lane    = t & 31;
  const int lane_hi = lane >> 4;
  const int lrow    = lane & 15;

#pragma unroll
  for (int pass = 0; pass < 2; ++pass) {
    const bf16_t* wrow[3];
    int col[3];
#pragma unroll
    for (int tt = 0; tt < 3; ++tt) {
      col[tt]  = (wave * 6 + pass * 3 + tt) * 16 + lrow;  // output column 0..767
      wrow[tt] = Wall + (size_t)col[tt] * D_IN;           // B[k][n] = W[col][k]
    }

    v8f acc[3] = {};
    v16bf b0[3], b1[3];
#pragma unroll
    for (int tt = 0; tt < 3; ++tt) b0[tt] = load_frag_contig(wrow[tt], 0, lane_hi);

    for (int kk = 0; kk < D_IN; kk += 64) {
      v16bf a0 = load_frag_contig(&xn[lrow][0], kk, lane_hi);
#pragma unroll
      for (int tt = 0; tt < 3; ++tt) b1[tt] = load_frag_contig(wrow[tt], kk + 32, lane_hi);
#pragma unroll
      for (int tt = 0; tt < 3; ++tt) acc[tt] = wmma_bf16(a0, b0[tt], acc[tt]);

      v16bf a1 = load_frag_contig(&xn[lrow][0], kk + 32, lane_hi);
      const int nk = (kk + 64 < D_IN) ? (kk + 64) : 0;    // branchless tail
#pragma unroll
      for (int tt = 0; tt < 3; ++tt) b0[tt] = load_frag_contig(wrow[tt], nk, lane_hi);
#pragma unroll
      for (int tt = 0; tt < 3; ++tt) acc[tt] = wmma_bf16(a1, b1[tt], acc[tt]);
    }

#pragma unroll
    for (int tt = 0; tt < 3; ++tt) {
#pragma unroll
      for (int r = 0; r < 8; ++r) {
        const int M    = r + (lane_hi ? 8 : 0);
        const int s    = m0 + M;
        const float vv = acc[tt][r];
        const int c    = col[tt];
        if (c < H_DIM)            Q[(size_t)(b * S_DIM + s) * H_DIM + c]            = (bf16_t)vv;
        else if (c < 2 * H_DIM)   K[(size_t)(b * S_DIM + s) * H_DIM + (c - H_DIM)]  = (bf16_t)vv;
        else                      Vt[((size_t)b * D_OUT + (c - 2 * H_DIM)) * S_DIM + s] = (bf16_t)vv;
      }
    }
  }
}

// ---------------- kernel 2: flash attention (bf16 WMMA, fp32 accumulate) ----------------
// grid (S/16, B), block 256 (8 waves). Block: 16 query rows; keys in chunks of 128.
// Per chunk: each wave computes one 16x16 score tile (4 WMMAs), all 256 threads
// run the online softmax, then each wave does 4 N-tiles x 4 K-steps of PV WMMA
// with ping-pong V-fragment buffers so loads overlap the WMMA chains.
__global__ void __launch_bounds__(256, 1)
mha_attn(const bf16_t* __restrict__ Q, const bf16_t* __restrict__ K,
         const bf16_t* __restrict__ Vt, float* __restrict__ out) {
  __shared__ __align__(16) float  scoreS[16][KCHUNK];   // scaled scores, 8 KB
  __shared__ __align__(16) bf16_t scoreP[16][KCHUNK];   // exp(s - m_new) bf16, 4 KB
  __shared__ float mArr[16], lArr[16], alphaArr[16];

  const int b  = blockIdx.y;
  const int m0 = blockIdx.x * 16;
  const int t  = threadIdx.x;
  const int wave    = t >> 5;
  const int lane    = t & 31;
  const int lane_hi = lane >> 4;
  const int lrow    = lane & 15;
  const float scale = 0.044194173824159216f;            // 1/sqrt(512)

  if (t < 16) { mArr[t] = -1e30f; lArr[t] = 0.f; alphaArr[t] = 1.f; }

  // Q A-fragments (16 x 128) cached in registers, identical across waves.
  const bf16_t* qrow = Q + (size_t)(b * S_DIM + m0 + lrow) * H_DIM;
  v16bf qa[4];
  load_frag4(qa, qrow, lane_hi);

  // this wave's 16x64 fp32 output accumulator: cols wave*64 .. +63
  v8f acc[4] = {};
  // Vt rows (contiguous along keys) for this wave's 4 column tiles
  const bf16_t* vtrow[4];
#pragma unroll
  for (int nt = 0; nt < 4; ++nt)
    vtrow[nt] = Vt + ((size_t)b * D_OUT + wave * 64 + nt * 16 + lrow) * S_DIM;

  __syncthreads();

  for (int j = 0; j < S_DIM; j += KCHUNK) {
    // ---- scores: wave w owns keys [j + w*16, j + w*16 + 16) ----
    {
      const int key = j + wave * 16 + lrow;
      const bf16_t* krow = K + (size_t)(b * S_DIM + key) * H_DIM;
      v16bf kb[4];
      load_frag4(kb, krow, lane_hi);           // one clause of 8 b128 loads
      v8f sacc = {};
#pragma unroll
      for (int kk = 0; kk < 4; ++kk) sacc = wmma_bf16(qa[kk], kb[kk], sacc);
#pragma unroll
      for (int r = 0; r < 8; ++r)
        scoreS[r + lane_hi * 8][wave * 16 + lrow] = sacc[r] * scale;
      if (j + KCHUNK < S_DIM) {
        __builtin_prefetch(K + (size_t)(b * S_DIM + key + KCHUNK) * H_DIM, 0, 1);
        __builtin_prefetch(vtrow[0] + j + KCHUNK, 0, 1);
        __builtin_prefetch(vtrow[2] + j + KCHUNK, 0, 1);
      }
    }
    __syncthreads();

    // ---- online softmax: 16 threads per row, 8 columns per thread ----
    {
      const int row = t >> 4;          // 0..15
      const int i   = t & 15;
      float sv[8]; float mx = -1e30f;
#pragma unroll
      for (int c = 0; c < 8; ++c) { sv[c] = scoreS[row][i * 8 + c]; mx = fmaxf(mx, sv[c]); }
#pragma unroll
      for (int off = 1; off < 16; off <<= 1) mx = fmaxf(mx, __shfl_xor(mx, off, 32));
      const float mo = mArr[row];
      const float mn = fmaxf(mo, mx);
      const float al = __expf(mo - mn);
      float psum = 0.f;
#pragma unroll
      for (int c = 0; c < 8; ++c) {
        float p = __expf(sv[c] - mn);
        psum += p;
        scoreP[row][i * 8 + c] = (bf16_t)p;
      }
#pragma unroll
      for (int off = 1; off < 16; off <<= 1) psum += __shfl_xor(psum, off, 32);
      if (i == 0) {
        lArr[row] = al * lArr[row] + psum;
        mArr[row] = mn;
        alphaArr[row] = al;
      }
    }
    __syncthreads();

    // ---- O = alpha*O + P Vchunk : 4 N-tiles x 4 K-steps, ping-pong V buffers ----
    {
      v16bf pa[4];
      load_frag4(pa, &scoreP[lrow][0], lane_hi);
      float alph[8];
#pragma unroll
      for (int r = 0; r < 8; ++r) alph[r] = alphaArr[r + lane_hi * 8];

      v16bf vbA[4], vbB[4];
      load_frag4(vbA, vtrow[0] + j, lane_hi);

      // nt = 0: load tile 1 while computing tile 0
#pragma unroll
      for (int r = 0; r < 8; ++r) acc[0][r] *= alph[r];
      load_frag4(vbB, vtrow[1] + j, lane_hi);
#pragma unroll
      for (int kk = 0; kk < 4; ++kk) acc[0] = wmma_bf16(pa[kk], vbA[kk], acc[0]);

      // nt = 1
#pragma unroll
      for (int r = 0; r < 8; ++r) acc[1][r] *= alph[r];
      load_frag4(vbA, vtrow[2] + j, lane_hi);
#pragma unroll
      for (int kk = 0; kk < 4; ++kk) acc[1] = wmma_bf16(pa[kk], vbB[kk], acc[1]);

      // nt = 2
#pragma unroll
      for (int r = 0; r < 8; ++r) acc[2][r] *= alph[r];
      load_frag4(vbB, vtrow[3] + j, lane_hi);
#pragma unroll
      for (int kk = 0; kk < 4; ++kk) acc[2] = wmma_bf16(pa[kk], vbA[kk], acc[2]);

      // nt = 3
#pragma unroll
      for (int r = 0; r < 8; ++r) acc[3][r] *= alph[r];
#pragma unroll
      for (int kk = 0; kk < 4; ++kk) acc[3] = wmma_bf16(pa[kk], vbB[kk], acc[3]);
    }
    __syncthreads();   // protect scoreS/scoreP for next chunk
  }

  // ---- epilogue: divide by l, store fp32 ----
  float linv[8];
#pragma unroll
  for (int r = 0; r < 8; ++r) linv[r] = 1.0f / lArr[r + lane_hi * 8];
#pragma unroll
  for (int nt = 0; nt < 4; ++nt) {
#pragma unroll
    for (int r = 0; r < 8; ++r) {
      const int M   = r + (lane_hi ? 8 : 0);
      const int col = wave * 64 + nt * 16 + lrow;
      out[(size_t)(b * S_DIM + m0 + M) * D_OUT + col] = acc[nt][r] * linv[r];
    }
  }
}

// ---------------- host launch ----------------
extern "C" void kernel_launch(void* const* d_in, const int* in_sizes, int n_in,
                              void* d_out, int out_size, void* d_ws, size_t ws_size,
                              hipStream_t stream) {
  (void)in_sizes; (void)n_in; (void)out_size; (void)ws_size;
  const float* x    = (const float*)d_in[0];
  const float* ln_w = (const float*)d_in[1];
  const float* ln_b = (const float*)d_in[2];
  const float* Wq   = (const float*)d_in[3];
  const float* Wk   = (const float*)d_in[4];
  const float* Wv   = (const float*)d_in[5];
  float* out        = (float*)d_out;

  // workspace layout (bf16): Wall[768x512] | Q[B*S*128] | K[B*S*128] | Vt[B*512*S]
  char* ws = (char*)d_ws;
  const size_t wallB = (size_t)N_PROJ * D_IN * sizeof(bf16_t);         // 786432
  const size_t qkB   = (size_t)B_DIM * S_DIM * H_DIM * sizeof(bf16_t); // 8 MB
  bf16_t* Wall = (bf16_t*)ws;
  bf16_t* Qb   = (bf16_t*)(ws + wallB);
  bf16_t* Kb   = (bf16_t*)(ws + wallB + qkB);
  bf16_t* Vtb  = (bf16_t*)(ws + wallB + 2 * qkB);

  mha_cvt_w<<<(N_PROJ * D_IN + 255) / 256, 256, 0, stream>>>(Wq, Wk, Wv, Wall);

  dim3 grid(S_DIM / 16, B_DIM);
  mha_ln_proj<<<grid, 256, 0, stream>>>(x, ln_w, ln_b, Wall, Qb, Kb, Vtb);
  mha_attn<<<grid, 256, 0, stream>>>(Qb, Kb, Vtb, out);
}
